// DepthLoss_54631984005240
// MI455X (gfx1250) — compile-verified, matrix operand order checked by reading, not run on hardware
//
#include <hip/hip_runtime.h>

// Focal loss over depth (8,1,1536,2048) with bbox-union target mask.
// Mask built per 16x16 tile via v_wmma_f32_16x16x32_f16 (rank-64 indicator
// matmul, exact in f16/f32), fused with focal loss + reduction.

typedef __attribute__((ext_vector_type(16))) _Float16 v16h;
typedef __attribute__((ext_vector_type(8)))  float    v8f;

constexpr int BN = 8;          // batches
constexpr int HN = 1536;
constexpr int WN = 2048;
constexpr int NUM_GTS = 64;    // boxes (K dimension)
constexpr long NTOT = (long)BN * HN * WN;

__global__ void focal_init_kernel(float* out) { out[0] = 0.0f; }

__global__ __launch_bounds__(32)
void focal_wmma_kernel(const float* __restrict__ depth,
                       const int*   __restrict__ bbox,   // 64 x {tl_x, tl_y, br_x, br_y}
                       float*       __restrict__ out)
{
    __shared__ int4 boxes[NUM_GTS];

    const int lane = threadIdx.x;           // 0..31 (wave32)
    // Stage the 64 bbox descriptors in LDS (1 KB), one int4 per box.
    for (int i = lane; i < NUM_GTS; i += 32)
        boxes[i] = ((const int4*)bbox)[i];
    __syncthreads();

    const int tileX = blockIdx.x;           // 0..127  (W/16)
    const int tileY = blockIdx.y;           // 0..95   (H/16)
    const int m0 = tileY * 16;              // tile base row (h)
    const int n0 = tileX * 16;              // tile base col (w)
    const int hi = lane >> 4;               // lane half: 0 or 1
    const int lm = lane & 15;
    const int row = m0 + lm;                // A fragment: M = lane&15
    const int col = n0 + lm;                // B fragment: N = lane&15

    // counts(16x16) = rowInd(16x64) x colInd(64x16), two K=32 WMMA steps.
    v8f c = {};
#pragma unroll
    for (int step = 0; step < 2; ++step) {
        v16h a, b;
#pragma unroll
        for (int e = 0; e < 16; ++e) {
            // 16-bit A-fragment layout (ISA 7.12.2): slot e of lane-half hi -> K index
            const int k = (e & 7) + ((e >> 3) << 4) + (hi << 3) + step * 32;
            const int4 bx = boxes[k];
            // rows: h in [tl_y-1, br_y) ; cols: w in [tl_x-1, br_x)
            a[e] = (_Float16)((row >= bx.y - 1 && row < bx.w) ? 1.0f : 0.0f);
            b[e] = (_Float16)((col >= bx.x - 1 && col < bx.z) ? 1.0f : 0.0f);
        }
        c = __builtin_amdgcn_wmma_f32_16x16x32_f16(
                /*neg_a=*/false, a, /*neg_b=*/false, b,
                /*c_mod=*/(short)0, c, /*reuse_a=*/false, /*reuse_b=*/false);
    }

    // Fused focal loss over this tile for all 8 batches.
    // C/D layout: VGPR r holds (M = r + 8*hi, N = lane&15).
    float sum = 0.0f;
#pragma unroll
    for (int r = 0; r < 8; ++r) {
        const bool tgt = (c[r] > 0.5f);                 // counts > 0
        const int  h   = m0 + r + hi * 8;
        const long base = (long)h * WN + col;
#pragma unroll
        for (int bb = 0; bb < BN; ++bb) {
            const float p  = depth[(long)bb * ((long)HN * WN) + base];
            // log Z = m + log(1 + e^{-d}),  d = |1-2p|,  m = max(p, 1-p)
            const float d  = fabsf(1.0f - 2.0f * p);
            const float ed = __expf(-d);
            const float s  = 1.0f + ed;
            const float rs = __builtin_amdgcn_rcpf(s);
            const float ls = __logf(s);
            const float m  = fmaxf(p, 1.0f - p);
            const float xt = tgt ? (1.0f - p) : p;
            const float a_ = xt - m;                    // 0 or -d (exact)
            const float logpt = a_ - ls;
            const float pt = (a_ == 0.0f ? 1.0f : ed) * rs;
            const float om = 1.0f - pt;
            sum = fmaf(om * om, -logpt, sum);           // gamma = 2
        }
    }

    // Wave32 butterfly reduction, then one scaled atomic per wave.
#pragma unroll
    for (int off = 16; off > 0; off >>= 1)
        sum += __shfl_xor(sum, off, 32);
    if (lane == 0)
        atomicAdd(out, sum * (1.0f / (float)NTOT));     // LOSS_WEIGHT = 1
}

extern "C" void kernel_launch(void* const* d_in, const int* in_sizes, int n_in,
                              void* d_out, int out_size, void* d_ws, size_t ws_size,
                              hipStream_t stream)
{
    const float* depth = (const float*)d_in[0];   // (8,1,1536,2048) f32
    const int*   bbox  = (const int*)d_in[1];     // (64,4) int32
    float*       out   = (float*)d_out;           // scalar f32

    focal_init_kernel<<<1, 1, 0, stream>>>(out);

    dim3 grid(WN / 16, HN / 16);                  // 128 x 96 tiles, 1 wave each
    focal_wmma_kernel<<<grid, 32, 0, stream>>>(depth, bbox, out);
}
